// GAT_87308095193747
// MI455X (gfx1250) — compile-verified, matrix operand order checked by reading, not run on hardware
//
#include <hip/hip_runtime.h>
#include <hip/hip_bf16.h>
#include <math.h>

typedef __attribute__((ext_vector_type(16))) _Float16 v16h;
typedef __attribute__((ext_vector_type(8)))  float    v8f;

#define LEAKY_SLOPE 0.2f
#define BN_EPS 1e-5f

static inline long div_up(long a, long b) { return (a + b - 1) / b; }

// ---------------------------------------------------------------- utilities
__global__ __launch_bounds__(256) void gat_fill_f32(float* p, float v, long n) {
    long i = (long)blockIdx.x * blockDim.x + threadIdx.x;
    if (i < n) p[i] = v;
}

__global__ __launch_bounds__(256) void gat_f32_to_f16(const float* in, _Float16* out, long n) {
    long i = (long)blockIdx.x * blockDim.x + threadIdx.x;
    if (i < n) out[i] = (_Float16)in[i];
}

// W is [K][Nout] row-major f32; write Wt as [Nout][K] f16
__global__ __launch_bounds__(256) void gat_transpose_f16(const float* W, _Float16* Wt, int K, int Nout) {
    int i = blockIdx.x * blockDim.x + threadIdx.x;
    if (i < K * Nout) {
        int k = i / Nout, n = i % Nout;
        Wt[(long)n * K + k] = (_Float16)W[i];
    }
}

// ---------------------------------------------------------------- encoder / decoder (tiny K, scalar)
__global__ __launch_bounds__(256) void gat_encoder(const float* x, const float* W, const float* b,
                                                   float* z, int n, int fin, int enc) {
    long i = (long)blockIdx.x * blockDim.x + threadIdx.x;
    if (i >= (long)n * enc) return;
    int row = (int)(i / enc), col = (int)(i % enc);
    float acc = b[col];
    #pragma unroll
    for (int k = 0; k < 7; k++) acc += x[(long)row * fin + k] * W[(long)k * enc + col];
    z[i] = acc;
}

__global__ __launch_bounds__(256) void gat_decoder(const float* z, const float* W, const float* b,
                                                   float* out, int n, int din, int dout) {
    long i = (long)blockIdx.x * blockDim.x + threadIdx.x;
    if (i >= (long)n * dout) return;
    int row = (int)(i / dout), col = (int)(i % dout);
    float acc = b[col];
    for (int k = 0; k < din; k++) acc += z[(long)row * din + k] * W[(long)k * dout + col];
    out[i] = acc;
}

// ---------------------------------------------------------------- WMMA GEMM: C[M,Nout] = A[M,K] x B[K,Nout]
// A: f16 row-major [M][K];  Bt: f16 [Nout][K] (i.e. B transposed);  C: f32 row-major.
// One wave (32 lanes) computes one 16x16 tile, looping K in steps of 32.
__global__ __launch_bounds__(256) void gat_wmma_gemm(const _Float16* __restrict__ A,
                                                     const _Float16* __restrict__ Bt,
                                                     float* __restrict__ C,
                                                     int M, int K, int Nout) {
    long gtid = (long)blockIdx.x * blockDim.x + threadIdx.x;
    int lane = threadIdx.x & 31;
    long wave = gtid >> 5;
    int tilesN = Nout >> 4;
    long tm = wave / tilesN;
    int  tn = (int)(wave % tilesN);
    if (tm * 16 >= M) return;

    int hi    = lane >> 4;          // 0: lanes 0-15, 1: lanes 16-31
    int lsub  = lane & 15;
    int arow  = (int)tm * 16 + lsub;
    int bcol  = tn * 16 + lsub;
    int khalf = hi ? 8 : 0;         // per-lane k base per ISA A-layout

    v8f acc = {};
    for (int k0 = 0; k0 < K; k0 += 32) {
        const _Float16* ap = A  + (long)arow * K + k0 + khalf;
        const _Float16* bp = Bt + (long)bcol * K + k0 + khalf;
        v16h a, b;
        #pragma unroll
        for (int i = 0; i < 8; i++) { a[i]     = ap[i];      b[i]     = bp[i];      }
        #pragma unroll
        for (int i = 0; i < 8; i++) { a[8 + i] = ap[16 + i]; b[8 + i] = bp[16 + i]; }
        acc = __builtin_amdgcn_wmma_f32_16x16x32_f16(false, a, false, b,
                                                     (short)0, acc, false, false);
    }

    // D layout: VGPR j -> M = 16*tm + j + 8*hi, N = 16*tn + lsub
    int mbase = (int)tm * 16 + (hi ? 8 : 0);
    #pragma unroll
    for (int j = 0; j < 8; j++)
        C[(long)(mbase + j) * Nout + bcol] = acc[j];
}

// ---------------------------------------------------------------- attention coefficients per node
__global__ __launch_bounds__(256) void gat_attn_coef(const float* __restrict__ h,
                                                     const float* __restrict__ asrc,
                                                     const float* __restrict__ adst,
                                                     float* al_s, float* al_d,
                                                     int n, int H, int Cc) {
    long i = (long)blockIdx.x * blockDim.x + threadIdx.x;
    if (i >= (long)n * H) return;
    int node = (int)(i / H), hh = (int)(i % H);
    const float* hp = h + (long)node * H * Cc + (long)hh * Cc;
    float s = 0.f, d = 0.f;
    for (int c = 0; c < Cc; c++) {
        float v = hp[c];
        s += v * asrc[hh * Cc + c];
        d += v * adst[hh * Cc + c];
    }
    al_s[i] = s;
    al_d[i] = d;
}

// ---------------------------------------------------------------- edge helpers
__device__ __forceinline__ void edge_endpoints(const int* ei, long e, int E, int* s, int* d) {
    if (e < E) { *s = ei[e]; *d = ei[(long)E + e]; }
    else       { *s = (int)(e - E); *d = (int)(e - E); }   // self loops
}

__device__ __forceinline__ float leaky(float v) {
    return v > 0.f ? v : LEAKY_SLOPE * v;
}

__device__ __forceinline__ void atomicMaxF32(float* addr, float val) {
    if (val >= 0.f) atomicMax((int*)addr, __float_as_int(val));
    else            atomicMin((unsigned int*)addr, __float_as_uint(val));
}

// pass 1: segment max of e over dst
__global__ __launch_bounds__(256) void gat_edge_max(const int* __restrict__ ei, int E, int n, int H,
                                                    const float* __restrict__ al_s,
                                                    const float* __restrict__ al_d,
                                                    float* mbuf) {
    long i = (long)blockIdx.x * blockDim.x + threadIdx.x;
    long tot = (long)(E + n) * H;
    if (i >= tot) return;
    long e = i / H; int hh = (int)(i % H);
    int s, d; edge_endpoints(ei, e, E, &s, &d);
    float ev = leaky(al_s[(long)s * H + hh] + al_d[(long)d * H + hh]);
    atomicMaxF32(&mbuf[(long)d * H + hh], ev);
}

// pass 2: segment sum of exp(e - m[dst])
__global__ __launch_bounds__(256) void gat_edge_sum(const int* __restrict__ ei, int E, int n, int H,
                                                    const float* __restrict__ al_s,
                                                    const float* __restrict__ al_d,
                                                    const float* __restrict__ mbuf,
                                                    float* dbuf) {
    long i = (long)blockIdx.x * blockDim.x + threadIdx.x;
    long tot = (long)(E + n) * H;
    if (i >= tot) return;
    long e = i / H; int hh = (int)(i % H);
    int s, d; edge_endpoints(ei, e, E, &s, &d);
    float ev = leaky(al_s[(long)s * H + hh] + al_d[(long)d * H + hh]);
    float ex = expf(ev - mbuf[(long)d * H + hh]);
    atomicAdd(&dbuf[(long)d * H + hh], ex);
}

// pass 3: out[dst] += h[src] * alpha ; wave-per-edge, lanes stripe HC columns (coalesced 128B)
__global__ __launch_bounds__(256) void gat_edge_aggr(const int* __restrict__ ei, int E, int n,
                                                     int H, int Cc,
                                                     const float* __restrict__ al_s,
                                                     const float* __restrict__ al_d,
                                                     const float* __restrict__ mbuf,
                                                     const float* __restrict__ dbuf,
                                                     const float* __restrict__ h,
                                                     float* __restrict__ out) {
    long gtid = (long)blockIdx.x * blockDim.x + threadIdx.x;
    long e = gtid >> 5;
    int lane = threadIdx.x & 31;
    if (e >= (long)(E + n)) return;
    int s, d; edge_endpoints(ei, e, E, &s, &d);
    int HC = H * Cc;
    for (int j = lane; j < HC; j += 32) {
        int hh = j / Cc;
        float ev = leaky(al_s[(long)s * H + hh] + al_d[(long)d * H + hh]);
        float alpha = expf(ev - mbuf[(long)d * H + hh]) / (dbuf[(long)d * H + hh] + 1e-16f);
        atomicAdd(&out[(long)d * HC + j], h[(long)s * HC + j] * alpha);
    }
}

__global__ __launch_bounds__(256) void gat_bias_add(float* z, const float* b, long n, int cols) {
    long i = (long)blockIdx.x * blockDim.x + threadIdx.x;
    if (i < n) z[i] += b[i % cols];
}

// ---------------------------------------------------------------- batch norm
__global__ __launch_bounds__(256) void gat_bn_stats(const float* __restrict__ z, int n, int cols,
                                                    float* mean, float* invstd) {
    __shared__ float ssum[256];
    __shared__ float ssq[256];
    int col = blockIdx.x;
    float s = 0.f, q = 0.f;
    for (int r = threadIdx.x; r < n; r += blockDim.x) {
        float v = z[(long)r * cols + col];
        s += v; q += v * v;
    }
    ssum[threadIdx.x] = s; ssq[threadIdx.x] = q;
    __syncthreads();
    for (int st = 128; st > 0; st >>= 1) {
        if ((int)threadIdx.x < st) {
            ssum[threadIdx.x] += ssum[threadIdx.x + st];
            ssq[threadIdx.x]  += ssq[threadIdx.x + st];
        }
        __syncthreads();
    }
    if (threadIdx.x == 0) {
        float mu  = ssum[0] / (float)n;
        float var = ssq[0] / (float)n - mu * mu;
        mean[col]   = mu;
        invstd[col] = rsqrtf(var + BN_EPS);
    }
}

__global__ __launch_bounds__(256) void gat_bn_relu(float* z, const float* mean, const float* invstd,
                                                   const float* gamma, const float* beta,
                                                   long n, int cols) {
    long i = (long)blockIdx.x * blockDim.x + threadIdx.x;
    if (i >= n) return;
    int c = (int)(i % cols);
    float v = gamma[c] * (z[i] - mean[c]) * invstd[c] + beta[c];
    z[i] = v > 0.f ? v : 0.f;
}

// ---------------------------------------------------------------- host orchestration
extern "C" void kernel_launch(void* const* d_in, const int* in_sizes, int n_in,
                              void* d_out, int out_size, void* d_ws, size_t ws_size,
                              hipStream_t stream) {
    const int Nn  = in_sizes[0] / 7;        // 100000
    const int Ee  = in_sizes[1] / 2;        // 1600000
    const int ENC = 64, HEADS = 4, HID = 32, HC = 128, DEC_IN = 64, OUTD = 4;

    const float* x        = (const float*)d_in[0];
    const int*   ei       = (const int*)d_in[1];
    const float* enc_W    = (const float*)d_in[2];
    const float* enc_b    = (const float*)d_in[3];
    const float* in_W     = (const float*)d_in[4];
    const float* in_as    = (const float*)d_in[5];
    const float* in_ad    = (const float*)d_in[6];
    const float* in_b     = (const float*)d_in[7];
    const float* hid_W    = (const float*)d_in[8];
    const float* hid_as   = (const float*)d_in[9];
    const float* hid_ad   = (const float*)d_in[10];
    const float* hid_b    = (const float*)d_in[11];
    const float* bn_gamma = (const float*)d_in[12];
    const float* bn_beta  = (const float*)d_in[13];
    const float* out_W    = (const float*)d_in[14];
    const float* out_as   = (const float*)d_in[15];
    const float* out_ad   = (const float*)d_in[16];
    const float* out_b    = (const float*)d_in[17];
    const float* dec_W    = (const float*)d_in[18];
    const float* dec_b    = (const float*)d_in[19];

    // workspace carve-up
    size_t off = 0;
    auto carve = [&](size_t bytes) -> void* {
        void* p = (char*)d_ws + off;
        off += (bytes + 255) & ~(size_t)255;
        return p;
    };
    float*    zA   = (float*)   carve((size_t)Nn * HC * 4);
    float*    zB   = (float*)   carve((size_t)Nn * HC * 4);
    float*    hbuf = (float*)   carve((size_t)Nn * HC * 4);
    _Float16* zh   = (_Float16*)carve((size_t)Nn * HC * 2);
    float*    alS  = (float*)   carve((size_t)Nn * HEADS * 4);
    float*    alD  = (float*)   carve((size_t)Nn * HEADS * 4);
    float*    mbuf = (float*)   carve((size_t)Nn * HEADS * 4);
    float*    dbuf = (float*)   carve((size_t)Nn * HEADS * 4);
    float*    bnM  = (float*)   carve(HC * 4);
    float*    bnI  = (float*)   carve(HC * 4);
    _Float16* wtIn = (_Float16*)carve((size_t)ENC * HC * 2);
    _Float16* wtH0 = (_Float16*)carve((size_t)HC * HC * 2);
    _Float16* wtH1 = (_Float16*)carve((size_t)HC * HC * 2);
    _Float16* wtOut= (_Float16*)carve((size_t)HC * DEC_IN * 2);
    (void)ws_size;

    const int TB = 256;

    // weight transposes to f16 (once per launch)
    gat_transpose_f16<<<div_up((long)ENC * HC, TB), TB, 0, stream>>>(in_W, wtIn, ENC, HC);
    gat_transpose_f16<<<div_up((long)HC * HC, TB), TB, 0, stream>>>(hid_W, wtH0, HC, HC);
    gat_transpose_f16<<<div_up((long)HC * HC, TB), TB, 0, stream>>>(hid_W + (long)HC * HC, wtH1, HC, HC);
    gat_transpose_f16<<<div_up((long)HC * DEC_IN, TB), TB, 0, stream>>>(out_W, wtOut, HC, DEC_IN);

    // encoder: zA[N,64]
    gat_encoder<<<div_up((long)Nn * ENC, TB), TB, 0, stream>>>(x, enc_W, enc_b, zA, Nn, 7, ENC);

    // one GAT layer
    auto gat_layer = [&](const float* z_in, int K, const _Float16* Wt, int H, int Cc,
                         const float* att_s, const float* att_d, const float* bias, float* z_out) {
        int Nout = H * Cc;
        long cnt = (long)Nn * K;
        gat_f32_to_f16<<<div_up(cnt, TB), TB, 0, stream>>>(z_in, zh, cnt);

        long waves = (long)(Nn / 16) * (Nout / 16);
        gat_wmma_gemm<<<div_up(waves * 32, TB), TB, 0, stream>>>(zh, Wt, hbuf, Nn, K, Nout);

        gat_attn_coef<<<div_up((long)Nn * H, TB), TB, 0, stream>>>(hbuf, att_s, att_d, alS, alD, Nn, H, Cc);

        gat_fill_f32<<<div_up((long)Nn * H, TB), TB, 0, stream>>>(mbuf, -INFINITY, (long)Nn * H);
        gat_fill_f32<<<div_up((long)Nn * H, TB), TB, 0, stream>>>(dbuf, 0.f, (long)Nn * H);
        gat_fill_f32<<<div_up((long)Nn * Nout, TB), TB, 0, stream>>>(z_out, 0.f, (long)Nn * Nout);

        long tot = (long)(Ee + Nn) * H;
        gat_edge_max<<<div_up(tot, TB), TB, 0, stream>>>(ei, Ee, Nn, H, alS, alD, mbuf);
        gat_edge_sum<<<div_up(tot, TB), TB, 0, stream>>>(ei, Ee, Nn, H, alS, alD, mbuf, dbuf);

        long wav2 = (long)(Ee + Nn);
        gat_edge_aggr<<<div_up(wav2 * 32, TB), TB, 0, stream>>>(ei, Ee, Nn, H, Cc,
                                                               alS, alD, mbuf, dbuf, hbuf, z_out);
        gat_bias_add<<<div_up((long)Nn * Nout, TB), TB, 0, stream>>>(z_out, bias, (long)Nn * Nout, Nout);
    };

    auto bn_block = [&](float* z, const float* gamma, const float* beta) {
        gat_bn_stats<<<HC, TB, 0, stream>>>(z, Nn, HC, bnM, bnI);
        gat_bn_relu<<<div_up((long)Nn * HC, TB), TB, 0, stream>>>(z, bnM, bnI, gamma, beta,
                                                                  (long)Nn * HC, HC);
    };

    // layer 1: 64 -> 4x32
    gat_layer(zA, ENC, wtIn, HEADS, HID, in_as, in_ad, in_b, zB);
    bn_block(zB, bn_gamma + 0 * HC, bn_beta + 0 * HC);

    // hidden layer 0: 128 -> 4x32
    gat_layer(zB, HC, wtH0, HEADS, HID, hid_as, hid_ad, hid_b, zA);
    bn_block(zA, bn_gamma + 1 * HC, bn_beta + 1 * HC);

    // hidden layer 1
    gat_layer(zA, HC, wtH1, HEADS, HID, hid_as + HEADS * HID, hid_ad + HEADS * HID,
              hid_b + HC, zB);
    bn_block(zB, bn_gamma + 2 * HC, bn_beta + 2 * HC);

    // out layer: 128 -> 1x64, concat=False with 1 head == identity reshape
    gat_layer(zB, HC, wtOut, 1, DEC_IN, out_as, out_ad, out_b, zA);

    // decoder -> d_out [N,4]
    gat_decoder<<<div_up((long)Nn * OUTD, TB), TB, 0, stream>>>(zA, dec_W, dec_b,
                                                                (float*)d_out, Nn, DEC_IN, OUTD);
}
